// STDecoder1D_37443524887311
// MI455X (gfx1250) — matrix operand order repaired and checked by reading, not compiled, not run on hardware
//
#include <hip/hip_runtime.h>
#include <hip/hip_bf16.h>
#include <math.h>

typedef __bf16 bf16;
typedef __attribute__((ext_vector_type(16))) __bf16 v16bf;
typedef __attribute__((ext_vector_type(8)))  __bf16 v8bf;
typedef __attribute__((ext_vector_type(8)))  float  v8f;

#define T_TOK 16384   // B*N
#define NB    8       // batch
#define NTOK  2048    // tokens per batch

__device__ __forceinline__ float gelu_exact(float x) {
  return 0.5f * x * (1.0f + erff(x * 0.70710678118654752f));
}

// ---------------------------------------------------------------------------
// Batched WMMA GEMM: D = act(scale * A@B^T + bias + res)
//   A : [M x K] row-major bf16 (lda multiple of 8)
//   Bt: [N x K] row-major bf16 (= B transposed; ldb multiple of 8)
// One wave computes a 16x64 output strip (4 accumulators): per K-step the A
// fragment is loaded once and reused by 4 WMMAs -> 4 v_wmma per 10 b128 loads.
// K multiple of 32; N multiple of 64 (all shapes here satisfy this).
// ---------------------------------------------------------------------------
__global__ __launch_bounds__(32)
void wmma_gemm_bf16(const bf16* __restrict__ A, const bf16* __restrict__ Bt,
                    const float* __restrict__ bias, const float* __restrict__ res,
                    float* __restrict__ outF, bf16* __restrict__ outB,
                    int K, int lda, int ldb, int ldd, int ldres,
                    long long sA, long long sB, long long sD, long long sRes,
                    float scale, int act)
{
  const int lane = threadIdx.x;
  const int l    = lane & 15;
  const int hi   = lane >> 4;
  const int tn   = blockIdx.x << 6;   // 64-wide N strip
  const int tm   = blockIdx.y << 4;
  const int b    = blockIdx.z;

  const v8bf* Arow  = (const v8bf*)(A + (long long)b * sA + (long long)(tm + l) * lda);
  const bf16* Bbase = Bt + (long long)b * sB;

  v8f acc[4];
#pragma unroll
  for (int j = 0; j < 4; ++j) acc[j] = (v8f){0.f,0.f,0.f,0.f,0.f,0.f,0.f,0.f};

#pragma unroll 2
  for (int k0 = 0; k0 < K; k0 += 32) {
    const int c = k0 >> 3;
    // A 16x32 fragment: lane row tm+l, chunks [k0+8*hi .. +7], [k0+16+8*hi .. +7]
    v8bf a0 = Arow[c + hi];
    v8bf a1 = Arow[c + 2 + hi];
    v16bf av = __builtin_shufflevector(a0, a1, 0,1,2,3,4,5,6,7,8,9,10,11,12,13,14,15);
#pragma unroll
    for (int j = 0; j < 4; ++j) {
      // B 32x16 fragment (from B^T): lane col tn+16j+l, chunks [k0+8*hi .. +7], [k0+16+8*hi .. +7]
      const v8bf* Br = (const v8bf*)(Bbase + (long long)(tn + 16 * j + l) * ldb);
      v8bf b0 = Br[c + hi];
      v8bf b1 = Br[c + 2 + hi];
      v16bf bv = __builtin_shufflevector(b0, b1, 0,1,2,3,4,5,6,7,8,9,10,11,12,13,14,15);
      acc[j] = __builtin_amdgcn_wmma_f32_16x16x32_bf16(false, av, false, bv,
                                                       (short)0, acc[j], false, false);
    }
  }

#pragma unroll
  for (int j = 0; j < 4; ++j) {
    const int n = tn + 16 * j + l;
#pragma unroll
    for (int r = 0; r < 8; ++r) {
      int mr = tm + r + (hi << 3);      // C/D: VGPR r -> row tm + r + 8*hi
      float v = acc[j][r] * scale;
      if (bias) v += bias[n];
      if (res)  v += res[(long long)b * sRes + (long long)mr * ldres + n];
      if (act == 1) v = gelu_exact(v);
      long long o = (long long)b * sD + (long long)mr * ldd + n;
      if (outF) outF[o] = v;
      if (outB) outB[o] = (bf16)v;
    }
  }
}

// ---------------------------------------------------------------------------
// Tiled transpose + f32->bf16 cast:  dst[c*ldOut + r] = src[r*ldIn + inOff + c]
// ---------------------------------------------------------------------------
__global__ __launch_bounds__(256)
void transpose_cast(const float* __restrict__ src, bf16* __restrict__ dst,
                    int R, int C, int ldIn, long long inOff, long long sIn,
                    int ldOut, long long sOut)
{
  __shared__ float tile[32][33];
  const int b  = blockIdx.z;
  const int r0 = blockIdx.y * 32, c0 = blockIdx.x * 32;
  const int tx = threadIdx.x & 31, ty = threadIdx.x >> 5;
  const float* S = src + (long long)b * sIn + inOff;
#pragma unroll
  for (int j = 0; j < 32; j += 8) {
    int r = r0 + ty + j, c = c0 + tx;
    tile[ty + j][tx] = (r < R && c < C) ? S[(long long)r * ldIn + c] : 0.f;
  }
  __syncthreads();
  bf16* D = dst + (long long)b * sOut;
#pragma unroll
  for (int j = 0; j < 32; j += 8) {
    int c = c0 + ty + j, r = r0 + tx;
    if (c < C && r < R) D[(long long)c * ldOut + r] = (bf16)tile[tx][ty + j];
  }
}

__global__ __launch_bounds__(256)
void fill_zero_bf16(bf16* __restrict__ d, long long n) {
  long long i = (long long)blockIdx.x * 256 + threadIdx.x;
  if (i < n) d[i] = (bf16)0.0f;
}

__global__ __launch_bounds__(256)
void cvt_f32_bf16(const float* __restrict__ s, bf16* __restrict__ d, long long n) {
  long long i = (long long)blockIdx.x * 256 + threadIdx.x;
  if (i < n) d[i] = (bf16)s[i];
}

// ---------------------------------------------------------------------------
// Gaussian Fourier features: feat[t, c] = gelu(sin/cos(2*pi*(pos/16)*Bf[c%256]))
// ---------------------------------------------------------------------------
__global__ __launch_bounds__(256)
void coord_feat(const float* __restrict__ pos, const float* __restrict__ Bf,
                bf16* __restrict__ feat) {
  long long idx = (long long)blockIdx.x * 256 + threadIdx.x;  // t*512 + c
  int t = (int)(idx >> 9);
  int c = (int)(idx & 511);
  float p  = pos[t] * (1.0f / 16.0f);
  float xp = 6.2831853071795864f * p * Bf[c & 255];
  float v  = (c < 256) ? sinf(xp) : cosf(xp);
  feat[idx] = (bf16)gelu_exact(v);
}

// ---------------------------------------------------------------------------
// Fused inst-norm (256 feats, biased var, eps=1e-5) + rotary embedding.
// transOut=0: Y[t*yts + yoff + i]
// transOut=1: Y[(t>>11)*yBS + i*2048 + (t&2047)]   (k stored [dim, token])
// ---------------------------------------------------------------------------
__global__ __launch_bounds__(256)
void instnorm_rot(const float* __restrict__ X, int xts, int xoff,
                  bf16* __restrict__ Y, int yts, int yoff,
                  long long yBS, int transOut,
                  const float* __restrict__ pos, float mult)
{
  __shared__ float red[256];
  __shared__ float xn[256];
  int t = blockIdx.x, i = threadIdx.x;
  float v = X[(long long)t * xts + xoff + i];

  red[i] = v; __syncthreads();
  for (int s = 128; s > 0; s >>= 1) { if (i < s) red[i] += red[i + s]; __syncthreads(); }
  float mu = red[0] * (1.0f / 256.0f); __syncthreads();

  float d = v - mu;
  red[i] = d * d; __syncthreads();
  for (int s = 128; s > 0; s >>= 1) { if (i < s) red[i] += red[i + s]; __syncthreads(); }
  float var = red[0] * (1.0f / 256.0f);

  float nv = d * rsqrtf(var + 1e-5f);
  xn[i] = nv; __syncthreads();

  float f  = pos[t] * mult * expf(-(float)(i & 127) * 0.0719557842f); // ln(1e4)/128
  float rh = (i < 128) ? -xn[i + 128] : xn[i - 128];
  float o  = nv * cosf(f) + rh * sinf(f);
  if (transOut) {
    int bb = t >> 11, n = t & 2047;
    Y[(long long)bb * yBS + (long long)i * 2048 + n] = (bf16)o;
  } else {
    Y[(long long)t * yts + yoff + i] = (bf16)o;
  }
}

// ---------------------------------------------------------------------------
// LayerNorm (eps=1e-5, affine). catPos: col256 = pos/16, cols 257..ldout-1 = 0.
// ---------------------------------------------------------------------------
__global__ __launch_bounds__(256)
void layernorm_cat(const float* __restrict__ X, const float* __restrict__ g,
                   const float* __restrict__ bb, bf16* __restrict__ Y,
                   int ldout, int catPos, const float* __restrict__ pos)
{
  __shared__ float red[256];
  int t = blockIdx.x, i = threadIdx.x;
  float v = X[(long long)t * 256 + i];

  red[i] = v; __syncthreads();
  for (int s = 128; s > 0; s >>= 1) { if (i < s) red[i] += red[i + s]; __syncthreads(); }
  float mu = red[0] * (1.0f / 256.0f); __syncthreads();

  float d = v - mu;
  red[i] = d * d; __syncthreads();
  for (int s = 128; s > 0; s >>= 1) { if (i < s) red[i] += red[i + s]; __syncthreads(); }
  float var = red[0] * (1.0f / 256.0f);

  float o = d * rsqrtf(var + 1e-5f) * g[i] + bb[i];
  Y[(long long)t * ldout + i] = (bf16)o;
  if (catPos) {
    if (i == 0) Y[(long long)t * ldout + 256] = (bf16)(pos[t] * (1.0f / 16.0f));
    else if (i < 32 && 256 + i < ldout) Y[(long long)t * ldout + 256 + i] = (bf16)0.0f;
  }
}

// ---------------------------------------------------------------------------
// Final head: u[t] = d2g[t,:] @ D3[128,1] + b3
// ---------------------------------------------------------------------------
__global__ __launch_bounds__(128)
void head_out(const bf16* __restrict__ d2, const float* __restrict__ D3,
              const float* __restrict__ b3, float* __restrict__ u)
{
  __shared__ float red[128];
  int t = blockIdx.x, i = threadIdx.x;
  red[i] = (float)d2[(long long)t * 128 + i] * D3[i];
  __syncthreads();
  for (int s = 64; s > 0; s >>= 1) { if (i < s) red[i] += red[i + s]; __syncthreads(); }
  if (i == 0) u[t] = red[0] + b3[0];
}

// ---------------------------------------------------------------------------
extern "C" void kernel_launch(void* const* d_in, const int* in_sizes, int n_in,
                              void* d_out, int out_size, void* d_ws, size_t ws_size,
                              hipStream_t stream)
{
  (void)in_sizes; (void)n_in; (void)out_size; (void)ws_size;

  const float* z      = (const float*)d_in[0];
  const float* pos    = (const float*)d_in[1];
  const float* Bf     = (const float*)d_in[2];
  const float* Wcoord = (const float*)d_in[3];
  const float* Wkv    = (const float*)d_in[4];
  const float* Woca   = (const float*)d_in[5];
  const float* boca   = (const float*)d_in[6];
  const float* Wf1    = (const float*)d_in[7];
  const float* bfb1   = (const float*)d_in[8];
  const float* Wf2    = (const float*)d_in[9];
  const float* bfb2   = (const float*)d_in[10];
  const float* lng    = (const float*)d_in[11];
  const float* lnb    = (const float*)d_in[12];
  const float* P1     = (const float*)d_in[13];
  const float* P2     = (const float*)d_in[14];
  const float* P3     = (const float*)d_in[15];
  const float* Wqkv   = (const float*)d_in[16];
  const float* Woda   = (const float*)d_in[17];
  const float* boda   = (const float*)d_in[18];
  const float* ln2g   = (const float*)d_in[19];
  const float* ln2b   = (const float*)d_in[20];
  const float* D1     = (const float*)d_in[21];
  const float* D2     = (const float*)d_in[22];
  const float* D3     = (const float*)d_in[23];
  const float* b3     = (const float*)d_in[24];

  float* u_out  = (float*)d_out;            // [B,1,N] flat = 16384
  float* zp_out = (float*)d_out + T_TOK;    // [B,N,256]

  // ---- workspace bump allocator ----
  size_t cur = 0;
  char*  wsb = (char*)d_ws;
  auto alloc = [&](size_t bytes) -> char* {
    char* p = wsb + cur;
    cur = (cur + bytes + 255) & ~(size_t)255;
    return p;
  };
  const size_t F = sizeof(float), H2 = sizeof(bf16);

  // transposed bf16 weights ([N x K] row-major)
  bf16* WcT   = (bf16*)alloc((size_t)256 * 512 * H2);
  bf16* WkvT  = (bf16*)alloc((size_t)2048 * 256 * H2);
  bf16* WocaT = (bf16*)alloc((size_t)256 * 1024 * H2);
  bf16* Wf1T  = (bf16*)alloc((size_t)256 * 256 * H2);
  bf16* Wf2T  = (bf16*)alloc((size_t)256 * 256 * H2);
  bf16* P1T   = (bf16*)alloc((size_t)256 * 288 * H2);   // K padded 257 -> 288
  bf16* P2T   = (bf16*)alloc((size_t)256 * 256 * H2);
  bf16* P3T   = (bf16*)alloc((size_t)256 * 256 * H2);
  bf16* WqkvT = (bf16*)alloc((size_t)768 * 256 * H2);
  bf16* WodaT = (bf16*)alloc((size_t)256 * 256 * H2);
  bf16* D1T   = (bf16*)alloc((size_t)256 * 256 * H2);
  bf16* D2T   = (bf16*)alloc((size_t)128 * 256 * H2);

  bf16*  z_b    = (bf16*) alloc((size_t)T_TOK * 256 * H2);
  bf16*  feat_b = (bf16*) alloc((size_t)T_TOK * 512 * H2);
  float* x_f    = (float*)alloc((size_t)T_TOK * 256 * F);
  bf16*  q_b    = (bf16*) alloc((size_t)T_TOK * 256 * H2);
  bf16*  kT_b   = (bf16*) alloc((size_t)NB * 4 * 256 * NTOK * H2); // [b][h][d][n]
  bf16*  vT_b   = (bf16*) alloc((size_t)NB * 4 * 256 * NTOK * H2); // [b][h][e][n]

  size_t kvMark = cur;                       // region reused by decode phase
  float* kv_f   = (float*)alloc((size_t)T_TOK * 2048 * F);

  float* dotsT_f = (float*)alloc((size_t)NB * 256 * 1024 * F);
  bf16*  dotsT_b = (bf16*) alloc((size_t)NB * 256 * 1024 * H2);
  float* Mf_f    = (float*)alloc((size_t)NB * 256 * 256 * F);
  bf16*  MfT_b   = (bf16*) alloc((size_t)NB * 256 * 256 * H2);
  float* x2_f    = (float*)alloc((size_t)T_TOK * 256 * F);
  bf16*  x2_b    = (bf16*) alloc((size_t)T_TOK * 256 * H2);
  bf16*  h1_b    = (bf16*) alloc((size_t)T_TOK * 256 * H2);
  float* x3_f    = (float*)alloc((size_t)T_TOK * 256 * F);
  bf16*  hcat_b  = (bf16*) alloc((size_t)T_TOK * 288 * H2);
  bf16*  p1_b    = (bf16*) alloc((size_t)T_TOK * 256 * H2);
  bf16*  p2_b    = (bf16*) alloc((size_t)T_TOK * 256 * H2);
  bf16*  zp_b    = (bf16*) alloc((size_t)T_TOK * 256 * H2);

  // decode phase overlays dead kv_f region
  size_t cur2 = kvMark;
  auto alloc2 = [&](size_t bytes) -> char* {
    char* p = wsb + cur2;
    cur2 = (cur2 + bytes + 255) & ~(size_t)255;
    return p;
  };
  float* qkv_f = (float*)alloc2((size_t)T_TOK * 768 * F);
  bf16*  q2_b  = (bf16*) alloc2((size_t)T_TOK * 256 * H2);
  bf16*  k2T_b = (bf16*) alloc2((size_t)NB * 256 * NTOK * H2);
  bf16*  v2T_b = (bf16*) alloc2((size_t)NB * 256 * NTOK * H2);
  float* d2T_f = (float*)alloc2((size_t)NB * 256 * 256 * F);
  bf16*  d2T_b = (bf16*) alloc2((size_t)NB * 256 * 256 * H2);
  float* M2_f  = (float*)alloc2((size_t)NB * 256 * 256 * F);
  bf16*  M2T_b = (bf16*) alloc2((size_t)NB * 256 * 256 * H2);
  float* zd_f  = (float*)alloc2((size_t)T_TOK * 256 * F);
  bf16*  h2_b  = (bf16*) alloc2((size_t)T_TOK * 256 * H2);
  bf16*  d1g_b = (bf16*) alloc2((size_t)T_TOK * 256 * H2);
  bf16*  d2g_b = (bf16*) alloc2((size_t)T_TOK * 128 * H2);

  // ---- helpers ----
  auto CV = [&](const float* s, bf16* d, long long n) {
    cvt_f32_bf16<<<dim3((unsigned)((n + 255) / 256)), dim3(256), 0, stream>>>(s, d, n);
  };
  auto TR = [&](const float* s, bf16* d, int R, int C, int ldIn, long long inOff,
                long long sIn, int ldOut, long long sOut, int batch) {
    dim3 grid((C + 31) / 32, (R + 31) / 32, batch);
    transpose_cast<<<grid, dim3(256), 0, stream>>>(s, d, R, C, ldIn, inOff, sIn, ldOut, sOut);
  };
  auto G = [&](const bf16* A, const bf16* Bt, const float* bias, const float* res,
               float* oF, bf16* oB, int M, int Nn, int K, int lda, int ldb,
               int ldd, int ldres, long long sA, long long sB, long long sD,
               long long sRes, float scale, int act, int batch) {
    dim3 grid(Nn / 64, M / 16, batch);
    wmma_gemm_bf16<<<grid, dim3(32), 0, stream>>>(
        A, Bt, bias, res, oF, oB, K, lda, ldb, ldd, ldres, sA, sB, sD, sRes, scale, act);
  };

  // ---- weight prep: transpose + cast (P1T zero-padded to K=288) ----
  fill_zero_bf16<<<dim3((256 * 288 + 255) / 256), dim3(256), 0, stream>>>(P1T, 256 * 288);
  TR(Wcoord, WcT,   512, 256,  256, 0, 0, 512, 0, 1);
  TR(Wkv,    WkvT,  256, 2048, 2048, 0, 0, 256, 0, 1);
  TR(Woca,   WocaT, 1024, 256, 256, 0, 0, 1024, 0, 1);
  TR(Wf1,    Wf1T,  256, 256,  256, 0, 0, 256, 0, 1);
  TR(Wf2,    Wf2T,  256, 256,  256, 0, 0, 256, 0, 1);
  TR(P1,     P1T,   257, 256,  256, 0, 0, 288, 0, 1);
  TR(P2,     P2T,   256, 256,  256, 0, 0, 256, 0, 1);
  TR(P3,     P3T,   256, 256,  256, 0, 0, 256, 0, 1);
  TR(Wqkv,   WqkvT, 256, 768,  768, 0, 0, 256, 0, 1);
  TR(Woda,   WodaT, 256, 256,  256, 0, 0, 256, 0, 1);
  TR(D1,     D1T,   256, 256,  256, 0, 0, 256, 0, 1);
  TR(D2,     D2T,   256, 128,  128, 0, 0, 256, 0, 1);
  CV(z, z_b, (long long)T_TOK * 256);

  // ---- 1. coordinate projection ----
  coord_feat<<<dim3((T_TOK * 512) / 256), dim3(256), 0, stream>>>(pos, Bf, feat_b);
  G(feat_b, WcT, nullptr, nullptr, x_f, nullptr,
    T_TOK, 256, 512, 512, 512, 256, 0, 0, 0, 0, 0, 1.0f, 0, 1);

  // ---- 2. kv = z @ W_kv ----
  G(z_b, WkvT, nullptr, nullptr, kv_f, nullptr,
    T_TOK, 2048, 256, 256, 256, 2048, 0, 0, 0, 0, 0, 1.0f, 0, 1);

  // ---- 3. inst_norm + rotary (q row-major; k written transposed [d, n]) ----
  instnorm_rot<<<dim3(T_TOK), dim3(256), 0, stream>>>(
      x_f, 256, 0, q_b, 256, 0, 0, 0, pos, 2048.0f);
  for (int h = 0; h < 4; ++h)
    instnorm_rot<<<dim3(T_TOK), dim3(256), 0, stream>>>(
        kv_f, 2048, h * 256, kT_b + (long long)h * 256 * NTOK, 0, 0,
        (long long)4 * 256 * NTOK, 1, pos, 2048.0f);
  // v transposed: [b][h*256+e][n]
  TR(kv_f, vT_b, NTOK, 1024, 2048, 1024, (long long)NTOK * 2048,
     NTOK, (long long)1024 * NTOK, NB);

  // ---- 4. dots_h = k_h^T v_h  -> dotsT[b][d, h*256+e] ----
  for (int h = 0; h < 4; ++h)
    G(kT_b + (long long)h * 256 * NTOK, vT_b + (long long)h * 256 * NTOK,
      nullptr, nullptr, dotsT_f + h * 256, nullptr,
      256, 256, NTOK, NTOK, NTOK, 1024, 0,
      (long long)4 * 256 * NTOK, (long long)4 * 256 * NTOK, (long long)256 * 1024, 0,
      1.0f, 0, NB);
  CV(dotsT_f, dotsT_b, (long long)NB * 256 * 1024);

  // ---- 5. Mfold[b] = dotsT[b] @ Wo_ca (fused head merge + out-proj) ----
  G(dotsT_b, WocaT, nullptr, nullptr, Mf_f, nullptr,
    256, 256, 1024, 1024, 1024, 256, 0,
    (long long)256 * 1024, 0, 65536, 0, 1.0f, 0, NB);
  TR(Mf_f, MfT_b, 256, 256, 256, 0, 65536, 256, 65536, NB);

  // ---- 6. x2 = (q @ Mfold)/n + bo_ca + x ----
  G(q_b, MfT_b, boca, x_f, x2_f, x2_b,
    NTOK, 256, 256, 256, 256, 256, 256,
    (long long)NTOK * 256, 65536, (long long)NTOK * 256, (long long)NTOK * 256,
    1.0f / (float)NTOK, 0, NB);

  // ---- 7. CrossFormer FFN ----
  G(x2_b, Wf1T, bfb1, nullptr, nullptr, h1_b,
    T_TOK, 256, 256, 256, 256, 256, 0, 0, 0, 0, 0, 1.0f, 1, 1);
  G(h1_b, Wf2T, bfb2, x2_f, x3_f, nullptr,
    T_TOK, 256, 256, 256, 256, 256, 256, 0, 0, 0, 0, 1.0f, 0, 1);

  // ---- 8. propagator: LN + concat(pos/16) (K padded to 288) + GELU MLP ----
  layernorm_cat<<<dim3(T_TOK), dim3(256), 0, stream>>>(x3_f, lng, lnb, hcat_b, 288, 1, pos);
  G(hcat_b, P1T, nullptr, nullptr, nullptr, p1_b,
    T_TOK, 256, 288, 288, 288, 256, 0, 0, 0, 0, 0, 1.0f, 1, 1);
  G(p1_b, P2T, nullptr, nullptr, nullptr, p2_b,
    T_TOK, 256, 256, 256, 256, 256, 0, 0, 0, 0, 0, 1.0f, 1, 1);
  G(p2_b, P3T, nullptr, x3_f, zp_out, zp_b,
    T_TOK, 256, 256, 256, 256, 256, 256, 0, 0, 0, 0, 1.0f, 0, 1);

  // ---- 9. decode attention ----
  G(zp_b, WqkvT, nullptr, nullptr, qkv_f, nullptr,
    T_TOK, 768, 256, 256, 256, 768, 0, 0, 0, 0, 0, 1.0f, 0, 1);
  instnorm_rot<<<dim3(T_TOK), dim3(256), 0, stream>>>(
      qkv_f, 768, 0, q2_b, 256, 0, 0, 0, pos, 2048.0f);
  instnorm_rot<<<dim3(T_TOK), dim3(256), 0, stream>>>(
      qkv_f, 768, 256, k2T_b, 0, 0, (long long)256 * NTOK, 1, pos, 2048.0f);
  TR(qkv_f, v2T_b, NTOK, 256, 768, 512, (long long)NTOK * 768,
     NTOK, (long long)256 * NTOK, NB);

  G(k2T_b, v2T_b, nullptr, nullptr, d2T_f, nullptr,
    256, 256, NTOK, NTOK, NTOK, 256, 0,
    (long long)256 * NTOK, (long long)256 * NTOK, 65536, 0, 1.0f, 0, NB);
  CV(d2T_f, d2T_b, (long long)NB * 256 * 256);

  G(d2T_b, WodaT, nullptr, nullptr, M2_f, nullptr,
    256, 256, 256, 256, 256, 256, 0, 65536, 0, 65536, 0, 1.0f, 0, NB);
  TR(M2_f, M2T_b, 256, 256, 256, 0, 65536, 256, 65536, NB);

  G(q2_b, M2T_b, boda, zp_out, zd_f, nullptr,
    NTOK, 256, 256, 256, 256, 256, 256,
    (long long)NTOK * 256, 65536, (long long)NTOK * 256, (long long)NTOK * 256,
    1.0f / (float)NTOK, 0, NB);

  // ---- 10. decode FFN + head ----
  layernorm_cat<<<dim3(T_TOK), dim3(256), 0, stream>>>(zd_f, ln2g, ln2b, h2_b, 256, 0, pos);
  G(h2_b, D1T, nullptr, nullptr, nullptr, d1g_b,
    T_TOK, 256, 256, 256, 256, 256, 0, 0, 0, 0, 0, 1.0f, 1, 1);
  G(d1g_b, D2T, nullptr, nullptr, nullptr, d2g_b,
    T_TOK, 128, 256, 256, 256, 128, 0, 0, 0, 0, 0, 1.0f, 1, 1);
  head_out<<<dim3(T_TOK), dim3(128), 0, stream>>>(d2g_b, D3, b3, u_out);
}
